// Sensate_72988674228301
// MI455X (gfx1250) — matrix operand order repaired and checked by reading, not compiled, not run on hardware
//
#include <hip/hip_runtime.h>
#include <hip/hip_bf16.h>
#include <math.h>

typedef __bf16 bf16_t;
typedef bf16_t v16bf __attribute__((ext_vector_type(16)));
typedef float  v8f   __attribute__((ext_vector_type(8)));

#define B_    1024
#define T_    9
#define S_    4
#define D_    300
#define V_    50000
#define KPAD  320
#define NSPLIT 125     // number of N-stripes
#define NCOLS  400     // columns per stripe (125*400 = 50000)

// ---------------------------------------------------------------------------
// K0: zero the scalar accumulator
// ---------------------------------------------------------------------------
__global__ void init_kernel(float* acc) {
    if (threadIdx.x < 16) acc[threadIdx.x] = 0.0f;
}

// ---------------------------------------------------------------------------
// K1: gather senses, gating softmax, pooled vector, entropy/orth/distill terms
//     one block of 128 threads per batch row
// ---------------------------------------------------------------------------
__global__ __launch_bounds__(128)
void gate_kernel(const int* __restrict__ center_pos,
                 const int* __restrict__ qids,
                 const float* __restrict__ bert,
                 const float* __restrict__ SE,
                 float* __restrict__ pooledF,      // [B][300]
                 bf16_t* __restrict__ pooledB,     // [B][320] zero padded
                 float* __restrict__ acc)
{
    const int b = blockIdx.x;
    const int tid = threadIdx.x;

    __shared__ float sums[16];   // 0-3 scores, 4-7 norms^2, 8-13 pair dots
    __shared__ float sdist;
    if (tid < 16) sums[tid] = 0.0f;
    if (tid == 0) sdist = 0.0f;
    __syncthreads();

    const int cp = center_pos[b];
    const int cw = qids[b * T_ + cp];

    // each thread owns up to 3 strided d-indices
    float cs[S_][3];
    float ctx[3];
    int   dIdx[3];
    int nd = 0;
    for (int d = tid; d < D_; d += 128) dIdx[nd++] = d;

    for (int j = 0; j < nd; ++j) {
        const int d = dIdx[j];
        float sAll = 0.0f;
        for (int t = 0; t < T_; ++t) {
            const int w = qids[b * T_ + t];
            const float* base = SE + (size_t)w * S_ * D_ + d;
            sAll += base[0] + base[D_] + base[2 * D_] + base[3 * D_];
        }
        const float* cb = SE + (size_t)cw * S_ * D_ + d;
        float sC = 0.0f;
        #pragma unroll
        for (int s = 0; s < S_; ++s) { cs[s][j] = cb[s * D_]; sC += cs[s][j]; }
        ctx[j] = (sAll - sC) * (1.0f / 32.0f);   // mean over (T-1)*S = 32
    }

    // 14 scalar block-reductions via LDS atomics
    float part[14];
    #pragma unroll
    for (int k = 0; k < 14; ++k) part[k] = 0.0f;
    for (int j = 0; j < nd; ++j) {
        #pragma unroll
        for (int s = 0; s < S_; ++s) {
            part[s]     += cs[s][j] * ctx[j];
            part[4 + s] += cs[s][j] * cs[s][j];
        }
        part[8]  += cs[0][j] * cs[1][j];
        part[9]  += cs[0][j] * cs[2][j];
        part[10] += cs[0][j] * cs[3][j];
        part[11] += cs[1][j] * cs[2][j];
        part[12] += cs[1][j] * cs[3][j];
        part[13] += cs[2][j] * cs[3][j];
    }
    #pragma unroll
    for (int k = 0; k < 14; ++k) atomicAdd(&sums[k], part[k]);
    __syncthreads();

    // softmax over 4 senses (redundant per thread)
    const float invSqrtD = 0.057735026918962576f;   // 1/sqrt(300)
    float sc[S_], p[S_];
    float m = -3.4e38f;
    #pragma unroll
    for (int s = 0; s < S_; ++s) { sc[s] = sums[s] * invSqrtD; m = fmaxf(m, sc[s]); }
    float Z = 0.0f;
    #pragma unroll
    for (int s = 0; s < S_; ++s) { p[s] = __expf(sc[s] - m); Z += p[s]; }
    #pragma unroll
    for (int s = 0; s < S_; ++s) p[s] /= Z;

    if (tid == 0) {
        // gating entropy
        float ent = 0.0f;
        #pragma unroll
        for (int s = 0; s < S_; ++s) ent -= p[s] * __logf(fmaxf(p[s], 1e-12f));
        // orthogonality among normalized center senses
        float ns[S_];
        #pragma unroll
        for (int s = 0; s < S_; ++s) ns[s] = fmaxf(sqrtf(sums[4 + s]), 1e-12f);
        float o = 0.0f;
        float d01 = sums[8]  / (ns[0] * ns[1]); o += d01 * d01;
        float d02 = sums[9]  / (ns[0] * ns[2]); o += d02 * d02;
        float d03 = sums[10] / (ns[0] * ns[3]); o += d03 * d03;
        float d12 = sums[11] / (ns[1] * ns[2]); o += d12 * d12;
        float d13 = sums[12] / (ns[1] * ns[3]); o += d13 * d13;
        float d23 = sums[13] / (ns[2] * ns[3]); o += d23 * d23;
        atomicAdd(acc, 0.01f * ent / (float)B_ + 0.1f * o / ((float)B_ * 6.0f));
    }

    // pooled vector + distillation MSE contribution
    float dloc = 0.0f;
    for (int j = 0; j < nd; ++j) {
        const int d = dIdx[j];
        float pv = p[0] * cs[0][j] + p[1] * cs[1][j] + p[2] * cs[2][j] + p[3] * cs[3][j];
        pooledF[(size_t)b * D_ + d] = pv;
        pooledB[(size_t)b * KPAD + d] = (bf16_t)pv;
        float diff = pv - bert[(size_t)b * D_ + d];
        dloc += diff * diff;
    }
    if (tid < KPAD - D_) pooledB[(size_t)b * KPAD + D_ + tid] = (bf16_t)0.0f;

    atomicAdd(&sdist, dloc);
    __syncthreads();
    if (tid == 0) atomicAdd(acc, 0.3f * sdist / ((float)B_ * (float)D_));
}

// ---------------------------------------------------------------------------
// K2: bf16 WMMA GEMM pooled @ W^T fused with online (max, sumexp) per row
//     block = 256 threads = 8 waves; wave w owns rows [bx*128 + 16w, +16)
//     blockIdx.y selects a 400-column stripe of V
// ---------------------------------------------------------------------------
__global__ __launch_bounds__(256)
void gemm_lse_kernel(const bf16_t* __restrict__ Abf,   // [B][KPAD]
                     const float*  __restrict__ W,     // [V][300]
                     float* __restrict__ partials)     // [B][NSPLIT][2]
{
    const int lane  = threadIdx.x & 31;
    const int wave  = threadIdx.x >> 5;
    const int m0    = blockIdx.x * 128 + wave * 16;
    const int nbase = blockIdx.y * NCOLS;
    const int row   = m0 + (lane & 15);
    const int khalf = lane >> 4;

    // A operand: 16x32 bf16 tiles, loop-invariant over N -> keep in registers.
    // lane L: M = L&15, K = khalf*16 + i within k-step (matches B)
    v16bf a[10];
    {
        const bf16_t* ap = Abf + (size_t)row * KPAD;
        #pragma unroll
        for (int ks = 0; ks < 10; ++ks) {
            const int kb = ks * 32 + khalf * 16;
            #pragma unroll
            for (int i = 0; i < 16; ++i) {
                const int k = kb + i;
                a[ks][i] = ap[k];
            }
        }
    }

    float rmax[8], rsum[8];
    #pragma unroll
    for (int r = 0; r < 8; ++r) { rmax[r] = -3.4e38f; rsum[r] = 0.0f; }

    const int col0 = nbase + (lane & 15);
    for (int t = 0; t < NCOLS / 16; ++t) {
        const int col = col0 + t * 16;
        const size_t wrow = (size_t)col * D_;
        v8f cacc = {0.f, 0.f, 0.f, 0.f, 0.f, 0.f, 0.f, 0.f};
        #pragma unroll
        for (int ks = 0; ks < 10; ++ks) {
            // B operand 32x16: lane = N (&15), K = khalf*16 + i within k-step
            const int kb = ks * 32 + khalf * 16;
            v16bf bv;
            if (ks < 9) {                         // fully in-bounds: kb+15 <= 287
                const float4* p4 = (const float4*)(W + wrow + kb);
                #pragma unroll
                for (int q = 0; q < 4; ++q) {
                    float4 f = p4[q];
                    bv[q * 4 + 0] = (bf16_t)f.x;
                    bv[q * 4 + 1] = (bf16_t)f.y;
                    bv[q * 4 + 2] = (bf16_t)f.z;
                    bv[q * 4 + 3] = (bf16_t)f.w;
                }
            } else {                              // K tail 288..319, pad with 0
                #pragma unroll
                for (int i = 0; i < 16; ++i) {
                    const int k = kb + i;
                    const int kc = (k < D_) ? k : 0;
                    float f = W[wrow + kc] * ((k < D_) ? 1.0f : 0.0f);
                    bv[i] = (bf16_t)f;
                }
            }
            cacc = __builtin_amdgcn_wmma_f32_16x16x32_bf16(
                false, a[ks], false, bv, (short)0, cacc, false, false);
        }
        // online (max,sumexp): lane holds column `col` for 8 rows (khalf*8 + r)
        #pragma unroll
        for (int r = 0; r < 8; ++r) {
            const float v  = cacc[r];
            const float nm = fmaxf(rmax[r], v);
            rsum[r] = rsum[r] * __expf(rmax[r] - nm) + __expf(v - nm);
            rmax[r] = nm;
        }
    }

    // merge across the 16 lanes of each half (they share the same row set)
    #pragma unroll
    for (int r = 0; r < 8; ++r) {
        for (int off = 1; off < 16; off <<= 1) {
            const float om = __shfl_xor(rmax[r], off, 32);
            const float os = __shfl_xor(rsum[r], off, 32);
            const float nm = fmaxf(rmax[r], om);
            rsum[r] = rsum[r] * __expf(rmax[r] - nm) + os * __expf(om - nm);
            rmax[r] = nm;
        }
    }
    if ((lane & 15) == 0) {
        #pragma unroll
        for (int r = 0; r < 8; ++r) {
            const int rr = m0 + khalf * 8 + r;
            float* pp = partials + ((size_t)rr * NSPLIT + blockIdx.y) * 2;
            pp[0] = rmax[r];
            pp[1] = rsum[r];
        }
    }
}

// ---------------------------------------------------------------------------
// K3: per-row combine of 125 stripe partials + picked-logit dot -> L_w2v
// ---------------------------------------------------------------------------
__global__ __launch_bounds__(128)
void combine_kernel(const float* __restrict__ partials,
                    const float* __restrict__ pooledF,
                    const float* __restrict__ W,
                    const int*   __restrict__ ctx_ids,
                    float* __restrict__ acc)
{
    const int b = blockIdx.x;
    const int tid = threadIdx.x;

    float m = -3.4e38f, s = 0.0f;
    for (int j = tid; j < NSPLIT; j += 128) {
        const float* pp = partials + ((size_t)b * NSPLIT + j) * 2;
        const float om = pp[0], os = pp[1];
        const float nm = fmaxf(m, om);
        s = s * __expf(m - nm) + os * __expf(om - nm);
        m = nm;
    }
    __shared__ float sm[128], ss[128];
    sm[tid] = m; ss[tid] = s;
    __syncthreads();
    for (int off = 64; off > 0; off >>= 1) {
        if (tid < off) {
            const float om = sm[tid + off], os = ss[tid + off];
            const float nm = fmaxf(sm[tid], om);
            ss[tid] = ss[tid] * __expf(sm[tid] - nm) + os * __expf(om - nm);
            sm[tid] = nm;
        }
        __syncthreads();
    }

    const int cid = ctx_ids[b];
    float dloc = 0.0f;
    for (int d = tid; d < D_; d += 128)
        dloc += pooledF[(size_t)b * D_ + d] * W[(size_t)cid * D_ + d];
    __shared__ float sdot[128];
    sdot[tid] = dloc;
    __syncthreads();
    for (int off = 64; off > 0; off >>= 1) {
        if (tid < off) sdot[tid] += sdot[tid + off];
        __syncthreads();
    }
    if (tid == 0) {
        const float logZ = sm[0] + __logf(ss[0]);
        atomicAdd(acc, (logZ - sdot[0]) / (float)B_);
    }
}

// ---------------------------------------------------------------------------
// K4: L2 regularization: sum of squares of 75M params (bandwidth-bound)
// ---------------------------------------------------------------------------
__global__ __launch_bounds__(256)
void l2_kernel(const float* __restrict__ SE,
               const float* __restrict__ OE,
               float* __restrict__ acc)
{
    const size_t i      = (size_t)blockIdx.x * 256 + threadIdx.x;
    const size_t stride = (size_t)gridDim.x * 256;
    const size_t nSE4 = (size_t)V_ * S_ * D_ / 4;   // 15,000,000 float4
    const size_t nOE4 = (size_t)V_ * D_ / 4;        //  3,750,000 float4
    const float4* a = (const float4*)SE;
    const float4* c = (const float4*)OE;
    float s = 0.0f;
    for (size_t j = i; j < nSE4; j += stride) {
        float4 v = a[j];
        s += v.x * v.x + v.y * v.y + v.z * v.z + v.w * v.w;
    }
    for (size_t j = i; j < nOE4; j += stride) {
        float4 v = c[j];
        s += v.x * v.x + v.y * v.y + v.z * v.z + v.w * v.w;
    }
    __shared__ float sd[256];
    sd[threadIdx.x] = s;
    __syncthreads();
    for (int off = 128; off > 0; off >>= 1) {
        if (threadIdx.x < off) sd[threadIdx.x] += sd[threadIdx.x + off];
        __syncthreads();
    }
    if (threadIdx.x == 0) atomicAdd(acc, 0.001f * sd[0]);
}

// ---------------------------------------------------------------------------
// K5: finalize scalar output
// ---------------------------------------------------------------------------
__global__ void final_kernel(const float* __restrict__ acc, float* __restrict__ out) {
    out[0] = acc[0];
}

// ---------------------------------------------------------------------------
extern "C" void kernel_launch(void* const* d_in, const int* in_sizes, int n_in,
                              void* d_out, int out_size, void* d_ws, size_t ws_size,
                              hipStream_t stream) {
    const int*   center_pos = (const int*)d_in[0];
    const int*   ctx_ids    = (const int*)d_in[1];
    const int*   qids       = (const int*)d_in[2];
    const float* bert       = (const float*)d_in[3];
    const float* SE         = (const float*)d_in[4];
    const float* OE         = (const float*)d_in[5];
    float* out = (float*)d_out;

    // workspace layout (256B aligned sections)
    char* ws = (char*)d_ws;
    float*  acc      = (float*)ws;                                    // 64 B
    float*  pooledF  = (float*)(ws + 256);                            // B*300*4
    bf16_t* pooledB  = (bf16_t*)(ws + 256 + (size_t)B_ * D_ * 4);     // B*320*2
    float*  partials = (float*)(ws + 256 + (size_t)B_ * D_ * 4
                                        + (size_t)B_ * KPAD * 2);     // B*125*2*4

    init_kernel<<<1, 16, 0, stream>>>(acc);
    gate_kernel<<<B_, 128, 0, stream>>>(center_pos, qids, bert, SE,
                                        pooledF, pooledB, acc);
    gemm_lse_kernel<<<dim3(B_ / 128, NSPLIT), 256, 0, stream>>>(pooledB, OE, partials);
    combine_kernel<<<B_, 128, 0, stream>>>(partials, pooledF, OE, ctx_ids, acc);
    l2_kernel<<<2048, 256, 0, stream>>>(SE, OE, acc);
    final_kernel<<<1, 1, 0, stream>>>(acc, out);
}
